// MoERouter_8761733284179
// MI455X (gfx1250) — matrix operand order repaired and checked by reading, not compiled, not run on hardware
//
#include <hip/hip_runtime.h>
#include <hip/hip_bf16.h>

#define NUM_EXPERTS 64
#define TOPK 2
#define HIDDEN 2048
#define NTOKENS 16384           // B*S = 4*4096
#define TPB_TOKENS 128          // tokens per block (8 waves x 16)
#define KCHUNK 128              // K staging chunk for gate_w in LDS
#define GPAD 8                  // bf16 pad per gate row (bank spread)
#define AUX_COEF 0.01f

typedef __attribute__((ext_vector_type(16))) __bf16 v16bf;
typedef __attribute__((ext_vector_type(8)))  __bf16 v8bf;
typedef __attribute__((ext_vector_type(4)))  __bf16 v4bf;
typedef __attribute__((ext_vector_type(8)))  float  v8f;

// ---------------- workspace zero ----------------
__global__ void moe_zero_ws(float* __restrict__ ws) {
    ws[threadIdx.x] = 0.0f;     // 128 floats: [0:64) count, [64:128) prob sum
}

// ---------------- main router kernel ----------------
__global__ __launch_bounds__(256) void moe_router_main(
    const float* __restrict__ hidden,   // [NTOKENS, HIDDEN]
    const float* __restrict__ gate,     // [NUM_EXPERTS, HIDDEN]
    float* __restrict__ out,            // [NTOKENS*2] probs | [NTOKENS*2] idx | [1] loss
    float* __restrict__ ws_cnt,         // [64]
    float* __restrict__ ws_psum)        // [64]
{
    __shared__ __attribute__((aligned(16))) __bf16 s_gate[NUM_EXPERTS][KCHUNK + GPAD];
    __shared__ float s_log[TPB_TOKENS][NUM_EXPERTS + 1];   // padded rows (bank-friendly)
    __shared__ float s_cnt[NUM_EXPERTS];

    const int tid   = threadIdx.x;
    const int lane  = tid & 31;
    const int wave  = tid >> 5;
    const int row16 = lane & 15;      // token row within M-tile / expert col within N-tile
    const int hl    = lane >> 4;      // lane half (K split per ISA layouts)
    const int tok0  = blockIdx.x * TPB_TOKENS;

    const float* hrow = hidden + (size_t)(tok0 + wave * 16 + row16) * HIDDEN;

    v8f acc[4] = {};   // 4 N-tiles of 16 experts, f32 accum

    for (int kc = 0; kc < HIDDEN; kc += KCHUNK) {
        // ---- stage gate_w[:, kc:kc+KCHUNK] into LDS as bf16 (whole block) ----
        __syncthreads();
        #pragma unroll
        for (int it = 0; it < (NUM_EXPERTS * (KCHUNK / 4)) / 256; ++it) {
            const int i  = it * 256 + tid;
            const int e  = i / (KCHUNK / 4);
            const int kk = (i % (KCHUNK / 4)) * 4;
            const float4 g = *(const float4*)(gate + (size_t)e * HIDDEN + kc + kk);
            v4bf b;
            b[0] = (__bf16)g.x; b[1] = (__bf16)g.y; b[2] = (__bf16)g.z; b[3] = (__bf16)g.w;
            *(v4bf*)&s_gate[e][kk] = b;
        }
        __syncthreads();

        if (kc + KCHUNK < HIDDEN)   // pull next hidden chunk toward the caches
            __builtin_prefetch(hrow + kc + KCHUNK + hl * 8, 0, 3);

        #pragma unroll
        for (int ks = 0; ks < KCHUNK; ks += 32) {
            const int kb = kc + ks;
            // ---- A tile 16x32 bf16 (ISA 16-bit A layout):
            // lane(hl,row16): elems 0..7 = K kb+hl*8.., elems 8..15 = K kb+16+hl*8..
            const float4 a0 = *(const float4*)(hrow + kb + hl * 8);
            const float4 a1 = *(const float4*)(hrow + kb + hl * 8 + 4);
            const float4 a2 = *(const float4*)(hrow + kb + 16 + hl * 8);
            const float4 a3 = *(const float4*)(hrow + kb + 16 + hl * 8 + 4);
            v16bf A;
            A[0]  = (__bf16)a0.x; A[1]  = (__bf16)a0.y; A[2]  = (__bf16)a0.z; A[3]  = (__bf16)a0.w;
            A[4]  = (__bf16)a1.x; A[5]  = (__bf16)a1.y; A[6]  = (__bf16)a1.z; A[7]  = (__bf16)a1.w;
            A[8]  = (__bf16)a2.x; A[9]  = (__bf16)a2.y; A[10] = (__bf16)a2.z; A[11] = (__bf16)a2.w;
            A[12] = (__bf16)a3.x; A[13] = (__bf16)a3.y; A[14] = (__bf16)a3.z; A[15] = (__bf16)a3.w;

            // ---- 4 B tiles 32x16 bf16 from LDS (lane holds col N=row16, K=hl*8+j and 16+hl*8+j)
            #pragma unroll
            for (int t = 0; t < 4; ++t) {
                const __bf16* gp = &s_gate[t * 16 + row16][ks + hl * 8];
                const v8bf blo = *(const v8bf*)gp;
                const v8bf bhi = *(const v8bf*)(gp + 16);
                const v16bf B = __builtin_shufflevector(blo, bhi,
                    0,1,2,3,4,5,6,7,8,9,10,11,12,13,14,15);
                acc[t] = __builtin_amdgcn_wmma_f32_16x16x32_bf16(
                    false, A, false, B, (short)0, acc[t], false, false);
            }
        }
    }

    // ---- scatter C to logits LDS (C layout: VGPR r -> M = r + 8*hl, lane&15 -> N)
    #pragma unroll
    for (int r = 0; r < 8; ++r) {
        const int trow = wave * 16 + hl * 8 + r;
        s_log[trow][ 0 + row16] = acc[0][r];
        s_log[trow][16 + row16] = acc[1][r];
        s_log[trow][32 + row16] = acc[2][r];
        s_log[trow][48 + row16] = acc[3][r];
    }
    if (tid < NUM_EXPERTS) s_cnt[tid] = 0.0f;
    __syncthreads();

    // ---- phase 2: per-token softmax + top-2 (one thread per token) ----
    if (tid < TPB_TOKENS) {
        float v[NUM_EXPERTS];
        #pragma unroll
        for (int e = 0; e < NUM_EXPERTS; ++e) v[e] = s_log[tid][e];

        float m = v[0];
        #pragma unroll
        for (int e = 1; e < NUM_EXPERTS; ++e) m = fmaxf(m, v[e]);

        float sum = 0.0f;
        float v1 = -3.402823466e38f, v2 = -3.402823466e38f;
        int   i1 = 0, i2 = 0;
        #pragma unroll
        for (int e = 0; e < NUM_EXPERTS; ++e) {
            const float p = __expf(v[e] - m);
            v[e] = p;
            sum += p;
            if (p > v1)      { v2 = v1; i2 = i1; v1 = p; i1 = e; }
            else if (p > v2) { v2 = p;  i2 = e; }
        }
        const float inv = 1.0f / sum;
        #pragma unroll
        for (int e = 0; e < NUM_EXPERTS; ++e) s_log[tid][e] = v[e] * inv;  // probs

        const int token = tok0 + tid;
        const float pn = 1.0f / (v1 + v2);           // == p1/(p1+p2) renorm
        out[token * 2 + 0] = v1 * pn;
        out[token * 2 + 1] = v2 * pn;
        out[NTOKENS * TOPK + token * 2 + 0] = (float)i1;
        out[NTOKENS * TOPK + token * 2 + 1] = (float)i2;
        atomicAdd(&s_cnt[i1], 1.0f);
        atomicAdd(&s_cnt[i2], 1.0f);
    }
    __syncthreads();

    // ---- per-expert partial sums -> global workspace ----
    if (tid < NUM_EXPERTS) {
        float cs = 0.0f;
        #pragma unroll 4
        for (int t = 0; t < TPB_TOKENS; ++t) cs += s_log[t][tid];
        atomicAdd(&ws_psum[tid], cs);
        atomicAdd(&ws_cnt[tid],  s_cnt[tid]);
    }
}

// ---------------- aux-loss finalize ----------------
__global__ void moe_loss(const float* __restrict__ ws_cnt,
                         const float* __restrict__ ws_psum,
                         float* __restrict__ out)
{
    __shared__ float s[NUM_EXPERTS];
    const int e = threadIdx.x;
    s[e] = ws_cnt[e] * ws_psum[e];
    __syncthreads();
    for (int off = 32; off > 0; off >>= 1) {
        if (e < off) s[e] += s[e + off];
        __syncthreads();
    }
    if (e == 0) {
        const float aux = (float)NUM_EXPERTS * s[0] /
                          ((float)NTOKENS * (float)NTOKENS);
        out[NTOKENS * TOPK * 2] = aux * AUX_COEF;
    }
}

extern "C" void kernel_launch(void* const* d_in, const int* in_sizes, int n_in,
                              void* d_out, int out_size, void* d_ws, size_t ws_size,
                              hipStream_t stream) {
    const float* hidden = (const float*)d_in[0];   // [4,4096,2048] f32
    const float* gate   = (const float*)d_in[1];   // [64,2048] f32
    float* out     = (float*)d_out;                // 65537 floats
    float* ws      = (float*)d_ws;
    float* ws_cnt  = ws;
    float* ws_psum = ws + NUM_EXPERTS;

    moe_zero_ws<<<1, 128, 0, stream>>>(ws);
    moe_router_main<<<NTOKENS / TPB_TOKENS, 256, 0, stream>>>(
        hidden, gate, out, ws_cnt, ws_psum);
    moe_loss<<<1, NUM_EXPERTS, 0, stream>>>(ws_cnt, ws_psum, out);
}
